// Model_39676907886475
// MI455X (gfx1250) — matrix operand order repaired and checked by reading, not compiled, not run on hardware
//
#include <hip/hip_runtime.h>

typedef __attribute__((ext_vector_type(16))) _Float16 v16h;
typedef __attribute__((ext_vector_type(8)))  _Float16 v8h;
typedef __attribute__((ext_vector_type(2)))  _Float16 v2h;
typedef __attribute__((ext_vector_type(8)))  float    v8f;

#define B_  4
#define H_  16
#define S_  2048
#define D_  64
#define BQ  128          // q rows per workgroup (8 waves x 16)
#define TK  32           // keys per tile
#define NTILE (S_ / TK)  // 64
#define KSTR 72          // LDS row stride (f16) for K tile  (rows 16B-aligned)
#define VSTR 40          // LDS row stride (f16) for V^T tile
#define PSTR 40          // LDS row stride (f16) for P staging

__device__ __forceinline__ v16h lds_ld16(const _Float16* p) {
  v8h lo = *reinterpret_cast<const v8h*>(p);
  v8h hi = *reinterpret_cast<const v8h*>(p + 8);
  return __builtin_shufflevector(lo, hi, 0,1,2,3,4,5,6,7,8,9,10,11,12,13,14,15);
}

// raw f32 max: v_max_num_f32 already implements IEEE maximumNumber (quiets
// NaNs) -- bypasses the compiler's canonicalize self-max companions on
// non-canonical (WMMA-produced) inputs.
__device__ __forceinline__ float fmax32(float a, float b) {
  float d;
  asm("v_max_num_f32 %0, %1, %2" : "=v"(d) : "v"(a), "v"(b));
  return d;
}

// in-place packed f16 max (no canonicalize, no extra movs)
__device__ __forceinline__ void pkmax_ip(v2h& a, v2h b) {
  asm("v_pk_max_num_f16 %0, %0, %1" : "+v"(a) : "v"(b));
}

// xor-butterfly step within each 16-lane row via v_permlane16_b32 (pure VALU,
// keeps the serial reduction chain off the high-latency LDS pipe)
__device__ __forceinline__ void plane16_max_ip(v2h& a, int sel0, int sel1) {
  int x = __builtin_bit_cast(int, a);
  int y = __builtin_amdgcn_permlane16(x, x, sel0, sel1, false, false);
  pkmax_ip(a, __builtin_bit_cast(v2h, y));
}

__global__ __launch_bounds__(256)
void fa_fwd(const float* __restrict__ q, const float* __restrict__ k,
            const float* __restrict__ v, const float* __restrict__ scale,
            float* __restrict__ out)
{
  __shared__ __align__(16) _Float16 sK [2][TK][KSTR];  // [key][dim]  f16
  __shared__ __align__(16) _Float16 sVt[2][D_][VSTR];  // [dim][key]  f16 (transposed)
  __shared__ __align__(16) _Float16 sP [8][16][PSTR];  // per-wave P staging (keys interleaved)

  const int tid  = threadIdx.x;
  const int wave = tid >> 5;
  const int lane = tid & 31;
  const int ln16 = lane & 15;
  const int lgrp = lane >> 4;            // 0 or 1 (half-wave group)

  const int bh  = blockIdx.y;
  const int q0b = blockIdx.x * BQ;
  const size_t base = (size_t)bh * S_ * D_;

  // fold 1/scale and log2(e) into Q itself: scores come out of the QK^T WMMA
  // already in the base-2 softmax domain (no per-score multiply in the loop)
  const float sc = 1.4426950408889634f / scale[0];

  // ---------------- Q fragments (A-layout, 16x32 f16, two head-dim chunks) --
  const int qrow = q0b + wave * 16 + ln16;     // A: row M = lane%16 (both groups)
  const float* qp = q + base + (size_t)qrow * D_;
  v16h qA[2];
#pragma unroll
  for (int c = 0; c < 2; ++c) {
#pragma unroll
    for (int e = 0; e < 16; ++e) {
      int d = 32 * c + e + (lgrp ? 16 : 0);
      qA[c][e] = (_Float16)(qp[d] * sc);
    }
  }

  // all-ones B matrix: one extra WMMA accumulates running row-sums (l) with
  // exactly the same alpha-rescale recurrence as O.
  v16h onesB;
#pragma unroll
  for (int e = 0; e < 16; ++e) onesB[e] = (_Float16)1.0f;

  // ---------------- softmax state + accumulators ----------------------------
  float mrow[8];
#pragma unroll
  for (int i = 0; i < 8; ++i) mrow[i] = -INFINITY;
  v8f acc[4] = {};   // O tile: 16 rows x 64 dims (4 C-layout 16x16 tiles)
  v8f accL  = {};    // running l (every column identical)

  // cooperative tile-load mapping: thread -> (key, 8 consecutive dims)
  const int tkey = tid >> 3;           // 0..31
  const int td0  = (tid & 7) << 3;     // 0,8,...,56
  const float* kp0 = k + base + (size_t)tkey * D_ + td0;
  const float* vp0 = v + base + (size_t)tkey * D_ + td0;

  // ---------------- prologue: tile 0 into buffer 0 --------------------------
  {
    float4 a0 = *(const float4*)(kp0);
    float4 a1 = *(const float4*)(kp0 + 4);
    float4 b0 = *(const float4*)(vp0);
    float4 b1 = *(const float4*)(vp0 + 4);
    float kv[8] = {a0.x,a0.y,a0.z,a0.w,a1.x,a1.y,a1.z,a1.w};
    float vv[8] = {b0.x,b0.y,b0.z,b0.w,b1.x,b1.y,b1.z,b1.w};
    v8h kh;
#pragma unroll
    for (int j = 0; j < 8; ++j) kh[j] = (_Float16)kv[j];
    *reinterpret_cast<v8h*>(&sK[0][tkey][td0]) = kh;
#pragma unroll
    for (int j = 0; j < 8; ++j) sVt[0][td0 + j][tkey] = (_Float16)vv[j];
  }
  __syncthreads();

  // ---------------- main loop over key tiles --------------------------------
  for (int it = 0; it < NTILE; ++it) {
    const int buf = it & 1;
    const bool havenext = (it + 1) < NTILE;

    // issue next tile's global loads now; latency hides under WMMA below
    float4 ka0, ka1, va0, va1;
    if (havenext) {
      const float* kp = kp0 + (size_t)(it + 1) * TK * D_;
      const float* vp = vp0 + (size_t)(it + 1) * TK * D_;
      ka0 = *(const float4*)(kp);
      ka1 = *(const float4*)(kp + 4);
      va0 = *(const float4*)(vp);
      va1 = *(const float4*)(vp + 4);
      if (it + 2 < NTILE) {
        __builtin_prefetch(kp0 + (size_t)(it + 2) * TK * D_, 0, 1);
        __builtin_prefetch(vp0 + (size_t)(it + 2) * TK * D_, 0, 1);
      }
    }

    // ---- scores: S = (Q*sc) @ K^T  (two 16-key subtiles, D=64 as two K=32) -
    v8f s0 = {}, s1 = {};
    {
      // B-layout: N(col)=key=lane%16, K(dim) elem e -> 32c + e + 16*lgrp
      const _Float16* kr0 = &sK[buf][ln16     ][lgrp * 16];
      const _Float16* kr1 = &sK[buf][16 + ln16][lgrp * 16];
      v16h b00 = lds_ld16(kr0);        // keys 0-15,  dims  0-31
      v16h b01 = lds_ld16(kr0 + 32);   // keys 0-15,  dims 32-63
      v16h b10 = lds_ld16(kr1);        // keys 16-31, dims  0-31
      v16h b11 = lds_ld16(kr1 + 32);
      s0 = __builtin_amdgcn_wmma_f32_16x16x32_f16(false, qA[0], false, b00, (short)0, s0, false, false);
      s0 = __builtin_amdgcn_wmma_f32_16x16x32_f16(false, qA[1], false, b01, (short)0, s0, false, false);
      s1 = __builtin_amdgcn_wmma_f32_16x16x32_f16(false, qA[0], false, b10, (short)0, s1, false, false);
      s1 = __builtin_amdgcn_wmma_f32_16x16x32_f16(false, qA[1], false, b11, (short)0, s1, false, false);
    }

    // ---- online softmax (base-2), C layout: row M = r + 8*lgrp, col = lane%16
    // packed-f16 butterfly max over the 16 lanes of each half-wave group.
    // f16 precision of the stabilizer m is exact for softmax: m cancels in O/l.
    v2h hm[4];
#pragma unroll
    for (int j = 0; j < 4; ++j) {
      float t0 = fmax32(s0[2*j],   s1[2*j]);
      float t1 = fmax32(s0[2*j+1], s1[2*j+1]);
      v2h h; h[0] = (_Float16)t0; h[1] = (_Float16)t1;
      hm[j] = h;
    }
    // xor lane-select tables for v_permlane16 (lanes 0-7 word, lanes 8-15 word)
    const int selLo[4] = {0x67452301, 0x54761032, 0x32107654, (int)0xFEDCBA98};
    const int selHi[4] = {(int)0xEFCDAB89, (int)0xDCFE98BA, (int)0xBA98FEDC, 0x76543210};
#pragma unroll
    for (int st = 0; st < 4; ++st) {
#pragma unroll
      for (int j = 0; j < 4; ++j)
        plane16_max_ip(hm[j], selLo[st], selHi[st]);
    }
    float rmax[8];
#pragma unroll
    for (int j = 0; j < 4; ++j) {
      rmax[2*j]   = (float)hm[j][0];
      rmax[2*j+1] = (float)hm[j][1];
    }

#pragma unroll
    for (int r = 0; r < 8; ++r) {
      float mn    = fmax32(mrow[r], rmax[r]);
      float alpha = __builtin_amdgcn_exp2f(mrow[r] - mn);
      float p0    = __builtin_amdgcn_exp2f(s0[r] - mn);
      float p1    = __builtin_amdgcn_exp2f(s1[r] - mn);
      mrow[r] = mn;
#pragma unroll
      for (int nt = 0; nt < 4; ++nt) acc[nt][r] *= alpha;
      accL[r] *= alpha;
      // interleaved store: col 2*n holds key n, col 2*n+1 holds key 16+n
      v2h pk; pk[0] = (_Float16)p0; pk[1] = (_Float16)p1;
      *reinterpret_cast<v2h*>(&sP[wave][r + lgrp * 8][2 * ln16]) = pk;
    }

    // same-wave LDS C->A transpose handoff: LDS is in-order per wave;
    // fence + dscnt wait for safety
    __builtin_amdgcn_wave_barrier();
    asm volatile("s_wait_dscnt 0x0" ::: "memory");

    // ---- O += P @ V ; l += P @ 1  ------------------------------------------
    {
      // A row M = lane%16; 16 contiguous interleaved f16 at col 16*lgrp
      const _Float16* pr = &sP[wave][ln16][16 * lgrp];
      v8h plo = *reinterpret_cast<const v8h*>(pr);
      v8h phi = *reinterpret_cast<const v8h*>(pr + 8);
      // de-interleave: e<8 -> even positions (keys 8g+e), e>=8 -> odd (16+8g+e-8)
      v16h pa = __builtin_shufflevector(plo, phi,
                  0,2,4,6,8,10,12,14,1,3,5,7,9,11,13,15);
#pragma unroll
      for (int nt = 0; nt < 4; ++nt) {
        // B-layout: N = dim = lane%16 + 16*nt ; K elem e<8 -> key 8*lgrp+e, e>=8 -> key 16+8*lgrp+(e-8)
        const _Float16* vr = &sVt[buf][nt * 16 + ln16][lgrp * 8];
        v16h vb = __builtin_shufflevector(*reinterpret_cast<const v8h*>(vr), *reinterpret_cast<const v8h*>(vr + 16), 0,1,2,3,4,5,6,7,8,9,10,11,12,13,14,15);
        acc[nt] = __builtin_amdgcn_wmma_f32_16x16x32_f16(false, pa, false, vb, (short)0, acc[nt], false, false);
      }
      accL = __builtin_amdgcn_wmma_f32_16x16x32_f16(false, pa, false, onesB, (short)0, accL, false, false);
    }

    // ---- store next tile to the other LDS buffer ---------------------------
    if (havenext) {
      float kv[8] = {ka0.x,ka0.y,ka0.z,ka0.w,ka1.x,ka1.y,ka1.z,ka1.w};
      float vv[8] = {va0.x,va0.y,va0.z,va0.w,va1.x,va1.y,va1.z,va1.w};
      v8h kh;
#pragma unroll
      for (int j = 0; j < 8; ++j) kh[j] = (_Float16)kv[j];
      *reinterpret_cast<v8h*>(&sK[buf ^ 1][tkey][td0]) = kh;
#pragma unroll
      for (int j = 0; j < 8; ++j) sVt[buf ^ 1][td0 + j][tkey] = (_Float16)vv[j];
    }
    __syncthreads();
  }

  // ---------------- epilogue: normalize rows and store ----------------------
#pragma unroll
  for (int r = 0; r < 8; ++r) {
    float inv_l = 1.0f / accL[r];   // every column of accL holds l(row)
    float* orow = out + base + (size_t)(q0b + wave * 16 + r + (lgrp ? 8 : 0)) * D_;
#pragma unroll
    for (int nt = 0; nt < 4; ++nt)
      orow[nt * 16 + ln16] = acc[nt][r] * inv_l;
  }
}

extern "C" void kernel_launch(void* const* d_in, const int* in_sizes, int n_in,
                              void* d_out, int out_size, void* d_ws, size_t ws_size,
                              hipStream_t stream) {
  const float* q     = (const float*)d_in[0];
  const float* k     = (const float*)d_in[1];
  const float* v     = (const float*)d_in[2];
  const float* scale = (const float*)d_in[3];
  float* out = (float*)d_out;
  (void)in_sizes; (void)n_in; (void)out_size; (void)d_ws; (void)ws_size;

  dim3 grid(S_ / BQ, B_ * H_);
  dim3 block(256);
  fa_fwd<<<grid, block, 0, stream>>>(q, k, v, scale, out);
}
